// MoQETorch_79517024518928
// MI455X (gfx1250) — matrix operand order repaired and checked by reading, not compiled, not run on hardware
//
#include <hip/hip_runtime.h>

// ---------------- problem constants ----------------
#define D_MODEL 1024
#define NBLK    16          // scale blocks per row (block width 64)
#define BT      2048        // B*T tokens
#define NLAYER  12
#define VDIM    50257
#define LDA     72          // padded LDS row stride (halfs)
#define LDB     72

typedef __attribute__((ext_vector_type(16))) _Float16 v16h;
typedef __attribute__((ext_vector_type(8)))  _Float16 v8h;
typedef __attribute__((ext_vector_type(4)))  _Float16 v4h;
typedef __attribute__((ext_vector_type(8)))  float    v8f;
typedef __attribute__((ext_vector_type(4)))  float    v4f;

static __device__ inline v8f vzero8() { v8f z = {0.f,0.f,0.f,0.f,0.f,0.f,0.f,0.f}; return z; }

// A fragment (16-bit A 16x32): lane m = lane&15, kh = lane>>4 selects K halves.
// lane holds K = {kh*8 .. kh*8+7} and {16+kh*8 .. 16+kh*8+7} of row m.
static __device__ inline v16h fragA(const _Float16* lds, int rowBase, int kk,
                                    int laneN, int laneH) {
  const _Float16* base = lds + (rowBase + laneN) * LDA + kk;
  v8h lo = *(const v8h*)(base + laneH * 8);
  v8h hi = *(const v8h*)(base + 16 + laneH * 8);
  return __builtin_shufflevector(lo, hi, 0,1,2,3,4,5,6,7,8,9,10,11,12,13,14,15);
}

// B fragment (16-bit B 32x16): lane col n = lane&15, kh = lane>>4,
// lane holds K = {kh*8 .. kh*8+7} and {16+kh*8 .. 16+kh*8+7} of column n.  LDS B tile is stored
// [n][k] (same as W's (out,in) row-major layout).
static __device__ inline v16h fragB(const _Float16* lds, int colBase, int kk,
                                    int laneN, int laneH) {
  const _Float16* base = lds + (colBase + laneN) * LDB + kk;
  v8h lo = *(const v8h*)(base + laneH * 8);
  v8h hi = *(const v8h*)(base + 16 + laneH * 8);
  return __builtin_shufflevector(lo, hi, 0,1,2,3,4,5,6,7,8,9,10,11,12,13,14,15);
}

static __device__ inline v8f wmma_f16(v16h a, v16h b, v8f c) {
  return __builtin_amdgcn_wmma_f32_16x16x32_f16(false, a, false, b, (short)0, c,
                                                false, false);
}

// ---------------- embedding gather: x = emb[ids], f32 + f16 copies ----------------
__global__ __launch_bounds__(256) void gather_kernel(const int* __restrict__ ids,
                                                     const float* __restrict__ emb,
                                                     float* __restrict__ xRes,
                                                     _Float16* __restrict__ xA) {
  int gid = blockIdx.x * 256 + threadIdx.x;     // one float4 per thread
  int e = gid * 4;
  int t = e >> 10;                               // D_MODEL == 1024
  int d = e & (D_MODEL - 1);
  int row = ids[t];
  v4f v = *(const v4f*)(emb + (size_t)row * D_MODEL + d);
  *(v4f*)(xRes + (size_t)t * D_MODEL + d) = v;
  v4h h = {(_Float16)v[0], (_Float16)v[1], (_Float16)v[2], (_Float16)v[3]};
  *(v4h*)(xA + (size_t)t * D_MODEL + d) = h;
}

// ---------------- router: p = sigmoid(x . rw + rb), one wave per token --------------
__global__ __launch_bounds__(256) void router_kernel(const float* __restrict__ x,
                                                     const float* __restrict__ rw,
                                                     const float* __restrict__ rb,
                                                     float* __restrict__ pbuf,
                                                     float* __restrict__ pout) {
  int wave = threadIdx.x >> 5;
  int lane = threadIdx.x & 31;
  int t = blockIdx.x * 8 + wave;
  float sum = 0.f;
#pragma unroll
  for (int j = 0; j < D_MODEL / 32; ++j) {
    int d = lane + j * 32;
    sum += x[(size_t)t * D_MODEL + d] * rw[d];
  }
#pragma unroll
  for (int off = 16; off > 0; off >>= 1) sum += __shfl_xor(sum, off, 32);
  if (lane == 0) {
    float p = 1.0f / (1.0f + __expf(-(sum + rb[0])));
    pbuf[t] = p;
    pout[t] = p;
  }
}

// ---------------- fused dual-expert layer GEMM -------------------------------------
// out0 = x @ dequant(w0,s0)^T ; out1 = x @ dequant(w1,s1)^T
// x_new = x + (1-p)*out0 + p*out1  -> f32 residual + f16 A-matrix for next layer
// 64x64 tile / 4 waves / wave does 32x32 per branch as 2x2 WMMA subtiles.
__global__ __launch_bounds__(128) void moqe_layer_kernel(
    const _Float16* __restrict__ xA, const float* __restrict__ xRes,
    const float* __restrict__ w0, const float* __restrict__ s0,
    const float* __restrict__ w1, const float* __restrict__ s1,
    const float* __restrict__ pbuf,
    float* __restrict__ xResOut, _Float16* __restrict__ xAOut) {
  __shared__ _Float16 ldsA[64 * LDA];
  __shared__ _Float16 ldsB0[64 * LDB];
  __shared__ _Float16 ldsB1[64 * LDB];

  const int tid  = threadIdx.x;
  const int wave = tid >> 5, lane = tid & 31;
  const int laneN = lane & 15, laneH = lane >> 4;
  const int wm = wave >> 1, wn = wave & 1;
  const int tileM = blockIdx.y * 64, tileN = blockIdx.x * 64;

  v8f acc0[4], acc1[4];
#pragma unroll
  for (int i = 0; i < 4; ++i) { acc0[i] = vzero8(); acc1[i] = vzero8(); }

  for (int kb = 0; kb < D_MODEL / 64; ++kb) {   // one scale block per iteration
    const int kBase = kb * 64;
    // stage A tile (f16, 64x64)
#pragma unroll
    for (int i = 0; i < 4; ++i) {
      int c = tid + i * 128;
      int row = c >> 3, cc = c & 7;
      *(v8h*)(ldsA + row * LDA + cc * 8) =
          *(const v8h*)(xA + (size_t)(tileM + row) * D_MODEL + kBase + cc * 8);
    }
    // stage B tiles with on-the-fly f32 -> f16 conversion (w is (out,in) row-major)
#pragma unroll
    for (int i = 0; i < 8; ++i) {
      int c = tid + i * 128;
      int row = c >> 4, cc = c & 15;
      size_t goff = (size_t)(tileN + row) * D_MODEL + kBase + cc * 4;
      v4f f0 = *(const v4f*)(w0 + goff);
      v4f f1 = *(const v4f*)(w1 + goff);
      v4h h0 = {(_Float16)f0[0], (_Float16)f0[1], (_Float16)f0[2], (_Float16)f0[3]};
      v4h h1 = {(_Float16)f1[0], (_Float16)f1[1], (_Float16)f1[2], (_Float16)f1[3]};
      *(v4h*)(ldsB0 + row * LDB + cc * 4) = h0;
      *(v4h*)(ldsB1 + row * LDB + cc * 4) = h1;
    }
    __syncthreads();

    // per-block, per-output-column dequant scales
    float sc0[2], sc1[2];
#pragma unroll
    for (int sn = 0; sn < 2; ++sn) {
      int colG = tileN + wn * 32 + sn * 16 + laneN;
      sc0[sn] = s0[(size_t)colG * NBLK + kb];
      sc1[sn] = s1[(size_t)colG * NBLK + kb];
    }

#pragma unroll
    for (int sm = 0; sm < 2; ++sm) {
      v16h aLo = fragA(ldsA, wm * 32 + sm * 16, 0,  laneN, laneH);
      v16h aHi = fragA(ldsA, wm * 32 + sm * 16, 32, laneN, laneH);
#pragma unroll
      for (int sn = 0; sn < 2; ++sn) {
        const int colB = wn * 32 + sn * 16;
        v8f t0 = vzero8(), t1 = vzero8();
        t0 = wmma_f16(aLo, fragB(ldsB0, colB, 0,  laneN, laneH), t0);
        t0 = wmma_f16(aHi, fragB(ldsB0, colB, 32, laneN, laneH), t0);
        t1 = wmma_f16(aLo, fragB(ldsB1, colB, 0,  laneN, laneH), t1);
        t1 = wmma_f16(aHi, fragB(ldsB1, colB, 32, laneN, laneH), t1);
        int idx = sm * 2 + sn;
        acc0[idx] += sc0[sn] * t0;   // f32 block-scale applied on accumulator
        acc1[idx] += sc1[sn] * t1;
      }
    }
    __syncthreads();
  }

  // epilogue: router combine + residual; write f32 residual and f16 next-layer A.
#pragma unroll
  for (int sm = 0; sm < 2; ++sm)
#pragma unroll
    for (int sn = 0; sn < 2; ++sn) {
      int idx = sm * 2 + sn;
      int nG = tileN + wn * 32 + sn * 16 + laneN;
#pragma unroll
      for (int i = 0; i < 8; ++i) {
        int mG = tileM + wm * 32 + sm * 16 + laneH * 8 + i;  // C layout: VGPR i -> M
        float pm = pbuf[mG];
        float xo = xRes[(size_t)mG * D_MODEL + nG];
        float v = xo + (1.0f - pm) * acc0[idx][i] + pm * acc1[idx][i];
        xResOut[(size_t)mG * D_MODEL + nG] = v;
        xAOut[(size_t)mG * D_MODEL + nG] = (_Float16)v;
      }
    }
}

// ---------------- final logits projection: (BT x D) x (V x D)^T -------------------
__global__ __launch_bounds__(128) void proj_kernel(const _Float16* __restrict__ xA,
                                                   const float* __restrict__ w,
                                                   float* __restrict__ out) {
  __shared__ _Float16 ldsA[64 * LDA];
  __shared__ _Float16 ldsB[64 * LDB];

  const int tid  = threadIdx.x;
  const int wave = tid >> 5, lane = tid & 31;
  const int laneN = lane & 15, laneH = lane >> 4;
  const int wm = wave >> 1, wn = wave & 1;
  const int tileM = blockIdx.y * 64, tileN = blockIdx.x * 64;

  v8f acc[4];
#pragma unroll
  for (int i = 0; i < 4; ++i) acc[i] = vzero8();

  for (int kb = 0; kb < D_MODEL / 64; ++kb) {
    const int kBase = kb * 64;
#pragma unroll
    for (int i = 0; i < 4; ++i) {
      int c = tid + i * 128;
      int row = c >> 3, cc = c & 7;
      *(v8h*)(ldsA + row * LDA + cc * 8) =
          *(const v8h*)(xA + (size_t)(tileM + row) * D_MODEL + kBase + cc * 8);
    }
#pragma unroll
    for (int i = 0; i < 8; ++i) {
      int c = tid + i * 128;
      int row = c >> 4, cc = c & 15;
      v4h h;
      if (tileN + row < VDIM) {
        v4f f = *(const v4f*)(w + (size_t)(tileN + row) * D_MODEL + kBase + cc * 4);
        h = (v4h){(_Float16)f[0], (_Float16)f[1], (_Float16)f[2], (_Float16)f[3]};
      } else {
        h = (v4h){(_Float16)0.f, (_Float16)0.f, (_Float16)0.f, (_Float16)0.f};
      }
      *(v4h*)(ldsB + row * LDB + cc * 4) = h;
    }
    __syncthreads();

#pragma unroll
    for (int sm = 0; sm < 2; ++sm) {
      v16h aLo = fragA(ldsA, wm * 32 + sm * 16, 0,  laneN, laneH);
      v16h aHi = fragA(ldsA, wm * 32 + sm * 16, 32, laneN, laneH);
#pragma unroll
      for (int sn = 0; sn < 2; ++sn) {
        const int colB = wn * 32 + sn * 16;
        int idx = sm * 2 + sn;
        acc[idx] = wmma_f16(aLo, fragB(ldsB, colB, 0,  laneN, laneH), acc[idx]);
        acc[idx] = wmma_f16(aHi, fragB(ldsB, colB, 32, laneN, laneH), acc[idx]);
      }
    }
    __syncthreads();
  }

#pragma unroll
  for (int sm = 0; sm < 2; ++sm)
#pragma unroll
    for (int sn = 0; sn < 2; ++sn) {
      int idx = sm * 2 + sn;
      int nG = tileN + wn * 32 + sn * 16 + laneN;
      if (nG < VDIM) {
#pragma unroll
        for (int i = 0; i < 8; ++i) {
          int mG = tileM + wm * 32 + sm * 16 + laneH * 8 + i;
          out[(size_t)mG * VDIM + nG] = acc[idx][i];
        }
      }
    }
}

// ---------------- driver ------------------------------------------------------------
extern "C" void kernel_launch(void* const* d_in, const int* in_sizes, int n_in,
                              void* d_out, int out_size, void* d_ws, size_t ws_size,
                              hipStream_t stream) {
  const int*   ids  = (const int*)d_in[0];
  const float* emb  = (const float*)d_in[1];
  const float* outw = (const float*)d_in[2];
  const float* w0   = (const float*)d_in[3];
  const float* s0   = (const float*)d_in[4];
  const float* w1   = (const float*)d_in[5];
  const float* s1   = (const float*)d_in[6];
  const float* rw   = (const float*)d_in[7];
  const float* rb   = (const float*)d_in[8];

  char* ws = (char*)d_ws;
  const size_t nx = (size_t)BT * D_MODEL;
  _Float16* xA0 = (_Float16*)ws;                       // 4 MB
  _Float16* xA1 = xA0 + nx;                            // 4 MB
  float* xR0 = (float*)(ws + 2 * nx * sizeof(_Float16)); // 8 MB
  float* xR1 = xR0 + nx;                               // 8 MB
  float* pbuf = xR1 + nx;                              // 8 KB

  float* logits    = (float*)d_out;
  float* routerOut = logits + (size_t)BT * VDIM;

  gather_kernel<<<(int)(nx / 4 / 256), 256, 0, stream>>>(ids, emb, xR0, xA0);

  for (int l = 0; l < NLAYER; ++l) {
    const _Float16* xAin = (l & 1) ? xA1 : xA0;
    const float*    xRin = (l & 1) ? xR1 : xR0;
    _Float16* xAout = (l & 1) ? xA0 : xA1;
    float*    xRout = (l & 1) ? xR0 : xR1;

    router_kernel<<<BT / 8, 256, 0, stream>>>(
        xRin, rw + (size_t)l * D_MODEL, rb + l, pbuf, routerOut + (size_t)l * BT);

    moqe_layer_kernel<<<dim3(D_MODEL / 64, BT / 64), 128, 0, stream>>>(
        xAin, xRin,
        w0 + (size_t)l * D_MODEL * D_MODEL, s0 + (size_t)l * D_MODEL * NBLK,
        w1 + (size_t)l * D_MODEL * D_MODEL, s1 + (size_t)l * D_MODEL * NBLK,
        pbuf, xRout, xAout);
  }

  proj_kernel<<<dim3((VDIM + 63) / 64, BT / 64), 128, 0, stream>>>(xA0, outw, logits);
}
